// SSDMobile_68530498175345
// MI455X (gfx1250) — matrix operand order repaired and checked by reading, not compiled, not run on hardware
//
#include <hip/hip_runtime.h>
#include <math.h>

typedef __attribute__((ext_vector_type(16))) _Float16 v16h;
typedef __attribute__((ext_vector_type(8)))  float    v8f;

#define BM 64
#define BN 32
#define BK 32
#define BKP 40   // LDS row stride in halves: 80 B, multiple of 16 -> b128-alignable

struct GemmP {
    const void*  X;      // input feature, NCHW (f32 or f16 per template)
    const float* W;      // weights [N][K] (1x1 conv: [Cout][Cin]), f32
    const float* scale;  // BN scale (mode 0)
    const float* shift;  // BN shift (mode 0) or conv bias (mode 1)
    _Float16*    Y;      // NCHW f16 feature out (mode 0)
    float*       out;    // final output base (mode 1)
    int M, N, K, HW;
    int mode;            // 0 = scale/shift + ReLU6 -> f16 NCHW ; 1 = +bias -> d_out scatter
    int na, ncpa, Aoff, col0, A;
};

struct StageRegs { float xv[8]; float wv[4]; };

// D = relu6(s*(X@W^T)+b) (f16 NCHW)  or  head scatter into d_out (f32).
// X[m,k] = X[((m/HW)*K + k)*HW + (m%HW)]   (1x1 conv over NCHW)
// Requires K % 8 == 0 (true for all uses: 80/960/512/256/128).
// Out-of-range M/N rows are loaded clamped (garbage) — only discarded D rows/cols
// are affected; K padding is explicitly zeroed.
template <typename TX>
__global__ __launch_bounds__(256) void ssdm_gemm_wmma(GemmP g) {
    __shared__ _Float16 Xs[2][BM][BKP];
    __shared__ _Float16 Ws[2][BN][BKP];

    const TX* __restrict__ Xp = (const TX*)g.X;

    const int t    = threadIdx.x;
    const int m0   = blockIdx.x * BM;
    const int n0   = blockIdx.y * BN;
    const int lane = t & 31;
    const int wave = t >> 5;     // 0..7
    const int wm   = wave >> 1;  // 0..3 (M tiles)
    const int wn   = wave & 1;   // 0..1 (N tiles)
    const int lr   = lane & 15;
    const int hi   = lane >> 4;

    // ---- X staging map: thread owns row xr, 8 consecutive K at xc0 (invariant over K loop)
    const int xr  = t & 63;
    const int xc0 = (t >> 6) << 3;                   // 0,8,16,24
    const int mc  = (m0 + xr < g.M) ? (m0 + xr) : (g.M - 1);   // clamp (garbage rows discarded)
    const int bb0 = mc / g.HW;
    const int ss0 = mc - bb0 * g.HW;
    const TX* __restrict__ xrow = Xp + ((long)bb0 * g.K) * g.HW + ss0;  // + k*HW

    // ---- W staging map: thread owns row wr (n), 4 consecutive K at wc0
    const int wr  = t & 31;
    const int wc0 = (t >> 5) << 2;                   // 0..28 step 4
    const int ncl = (n0 + wr < g.N) ? (n0 + wr) : (g.N - 1);   // clamp (garbage cols discarded)
    const float* __restrict__ wrow = g.W + (long)ncl * g.K;

    const long hw = g.HW;

    // ---- issue next tile's global loads (no LDS side effects)
    auto stage_load = [&](int k0) -> StageRegs {
        StageRegs r;
        if (k0 + xc0 < g.K) {     // group-uniform (K % 8 == 0)
            const TX* __restrict__ xp = xrow + (long)(k0 + xc0) * hw;
#pragma unroll
            for (int j = 0; j < 8; ++j) r.xv[j] = (float)xp[j * hw];
        } else {
#pragma unroll
            for (int j = 0; j < 8; ++j) r.xv[j] = 0.f;
        }
        int kw = k0 + wc0;
        if (kw < g.K) {           // group-uniform (K % 4 == 0)
            const float4 w4 = *reinterpret_cast<const float4*>(wrow + kw);
            r.wv[0] = w4.x; r.wv[1] = w4.y; r.wv[2] = w4.z; r.wv[3] = w4.w;
        } else {
            r.wv[0] = r.wv[1] = r.wv[2] = r.wv[3] = 0.f;
        }
        return r;
    };

    // ---- convert + store staged registers into LDS buffer pb
    auto stage_store = [&](const StageRegs& r, int pb) {
        union { _Float16 h[8]; uint4 u; } px;
#pragma unroll
        for (int j = 0; j < 8; ++j) px.h[j] = (_Float16)r.xv[j];
        *reinterpret_cast<uint4*>(&Xs[pb][xr][xc0]) = px.u;

        union { _Float16 h[4]; unsigned long long u; } pw;
#pragma unroll
        for (int j = 0; j < 4; ++j) pw.h[j] = (_Float16)r.wv[j];
        *reinterpret_cast<unsigned long long*>(&Ws[pb][wr][wc0]) = pw.u;
    };

    v8f acc = {0.f, 0.f, 0.f, 0.f, 0.f, 0.f, 0.f, 0.f};

    {
        StageRegs r0 = stage_load(0);
        stage_store(r0, 0);
    }
    __syncthreads();

    int pb = 0;
    for (int k0 = 0; k0 < g.K; k0 += BK) {
        const bool hasnext = (k0 + BK < g.K);
        StageRegs nxt;
        if (hasnext) nxt = stage_load(k0 + BK);   // global loads in flight...

        // ...while fragments are gathered and the WMMA runs
        v16h a, b;
#pragma unroll
        for (int v = 0; v < 8; ++v) {
            // A 16x32: K pair = (v/4)*16 + hi*8 + (v%4)*2
            int ka = ((v >> 2) << 4) + (hi << 3) + ((v & 3) << 1);
            // B 32x16: K pair = (v/4)*16 + hi*8 + (v%4)*2
            int kb = ((v >> 2) << 4) + (hi << 3) + ((v & 3) << 1);
            union { unsigned u; _Float16 h[2]; } pa, pc;
            pa.u = *reinterpret_cast<const unsigned*>(&Xs[pb][wm * 16 + lr][ka]);
            pc.u = *reinterpret_cast<const unsigned*>(&Ws[pb][wn * 16 + lr][kb]);
            a[2 * v] = pa.h[0]; a[2 * v + 1] = pa.h[1];
            b[2 * v] = pc.h[0]; b[2 * v + 1] = pc.h[1];
        }
        acc = __builtin_amdgcn_wmma_f32_16x16x32_f16(
            false, a, false, b, (short)0, acc, false, false);

        if (hasnext) stage_store(nxt, pb ^ 1);    // wait + cvt + LDS store after the MAC

        __syncthreads();
        pb ^= 1;
    }

    // ---- epilogue: D layout -> VGPR r: M = r + 8*hi, N = lr
#pragma unroll
    for (int r = 0; r < 8; ++r) {
        int gm = m0 + wm * 16 + hi * 8 + r;
        int gn = n0 + wn * 16 + lr;
        if (gm < g.M && gn < g.N) {
            float y = acc[r];
            int bb = gm / g.HW, s = gm - bb * g.HW;
            if (g.mode == 0) {
                y = y * g.scale[gn] + g.shift[gn];
                y = fminf(fmaxf(y, 0.f), 6.f);
                g.Y[((long)bb * g.N + gn) * g.HW + s] = (_Float16)y;
            } else {
                y += g.shift[gn];
                int aidx = gn / g.ncpa;
                int c    = gn - aidx * g.ncpa;
                long row = (long)bb * g.A + g.Aoff + (long)s * g.na + aidx;
                g.out[row * 95 + g.col0 + c] = y;
            }
        }
    }
}

// Depthwise 3x3 (pad 1) + BN-fold + ReLU6, f16 NCHW in/out, f32 math.
__global__ void ssdm_dwconv(const _Float16* __restrict__ X, const float* __restrict__ W9,
                            const float* __restrict__ scale, const float* __restrict__ shift,
                            _Float16* __restrict__ Y, int C, int Hin, int Hout, int stride,
                            int total) {
    int idx = blockIdx.x * blockDim.x + threadIdx.x;
    if (idx >= total) return;
    int Wout = Hout, Win = Hin;
    int ow = idx % Wout;
    int t1 = idx / Wout;
    int oh = t1 % Hout;
    int t2 = t1 / Hout;
    int c  = t2 % C;
    int bb = t2 / C;
    const _Float16* xp = X + ((long)bb * C + c) * Hin * Win;
    const float* wp = W9 + c * 9;
    float acc = 0.f;
#pragma unroll
    for (int ky = 0; ky < 3; ++ky) {
        int ih = oh * stride - 1 + ky;
        if (ih < 0 || ih >= Hin) continue;
#pragma unroll
        for (int kx = 0; kx < 3; ++kx) {
            int iw = ow * stride - 1 + kx;
            if (iw < 0 || iw >= Win) continue;
            acc += (float)xp[ih * Win + iw] * wp[ky * 3 + kx];
        }
    }
    float y = acc * scale[c] + shift[c];
    Y[((long)bb * C + c) * Hout * Wout + oh * Wout + ow] =
        (_Float16)fminf(fmaxf(y, 0.f), 6.f);
}

__device__ __forceinline__ float ssdm_c01(float v) { return fminf(fmaxf(v, 0.f), 1.f); }

// In-place prior decode + softmax + threshold over d_out [B, A, 95].
__global__ void ssdm_decode(float* __restrict__ out, int B, int A) {
    int idx = blockIdx.x * blockDim.x + threadIdx.x;
    if (idx >= B * A) return;
    int p = idx % A;

    const int offs[7] = {0, 1600, 2200, 2350, 2404, 2420, 2424};
    const int feat[6] = {20, 10, 5, 3, 2, 1};
    const int nas[6]  = {4, 6, 6, 6, 4, 4};
    int k = 0;
    while (k < 5 && p >= offs[k + 1]) ++k;
    int local = p - offs[k];
    int na = nas[k], f = feat[k];
    int s = local / na, a = local - s * na;
    int yy = s / f, xx = s - yy * f;

    float sk  = 0.07f + 0.176f * (float)k;
    float sk1 = (k == 5) ? 1.0f : 0.07f + 0.176f * (float)(k + 1);
    float skp = sqrtf(sk * sk1);
    float w, h;
    if (a == 0)      { w = sk;  h = sk; }
    else if (a == 1) { w = skp; h = skp; }
    else {
        int ai = (a - 2) >> 1;
        float r = sqrtf(ai == 0 ? 2.0f : 3.0f);
        if (((a - 2) & 1) == 0) { w = sk * r; h = sk / r; }
        else                    { w = sk / r; h = sk * r; }
    }
    float cx = (xx + 0.5f) / f, cy = (yy + 0.5f) / f;
    float x1 = ssdm_c01(cx - 0.5f * w), y1 = ssdm_c01(cy - 0.5f * h);
    float x2 = ssdm_c01(cx + 0.5f * w), y2 = ssdm_c01(cy + 0.5f * h);
    float pcx = 0.5f * (x1 + x2), pcy = 0.5f * (y1 + y2);
    float pw = x2 - x1, ph = y2 - y1;

    float* row = out + (long)idx * 95;
    float r0 = row[0], r1 = row[1], r2 = row[2], r3 = row[3];
    float dcx = r0 * 0.1f * pw + pcx;
    float dcy = r1 * 0.1f * ph + pcy;
    float dw  = expf(r2 * 0.2f) * pw;
    float dh  = expf(r3 * 0.2f) * ph;
    row[0] = ssdm_c01(dcx - 0.5f * dw);
    row[1] = ssdm_c01(dcy - 0.5f * dh);
    row[2] = ssdm_c01(dcx + 0.5f * dw);
    row[3] = ssdm_c01(dcy + 0.5f * dh);

    float mx = -3.4e38f;
    for (int c = 0; c < 91; ++c) mx = fmaxf(mx, row[4 + c]);
    float sum = 0.f;
    for (int c = 0; c < 91; ++c) sum += expf(row[4 + c] - mx);
    float inv = 1.f / sum;
    for (int c = 0; c < 91; ++c) {
        float sc = expf(row[4 + c] - mx) * inv;
        row[4 + c] = (sc > 0.05f) ? sc : 0.f;
    }
}

// ----------------------------- host side ------------------------------------

struct CbrP  { const float *w, *s, *b; };
struct HeadP { CbrP dw; const float *pw_w, *pw_b; };
struct NetP {
    const float *low, *high;
    CbrP proj_low, proj_high;
    CbrP edw[4], epw[4];
    HeadP cls[6], reg[6];
};

template <typename TX>
static void ssdm_run_gemm(const void* X, const float* W, const float* scale,
                          const float* shift, _Float16* Y, float* out,
                          int M, int N, int K, int HW, int mode,
                          int na, int ncpa, int Aoff, int col0, int A,
                          hipStream_t stream) {
    GemmP g{X, W, scale, shift, Y, out, M, N, K, HW, mode, na, ncpa, Aoff, col0, A};
    dim3 grid((M + BM - 1) / BM, (N + BN - 1) / BN);
    ssdm_gemm_wmma<TX><<<grid, 256, 0, stream>>>(g);
}

static void ssdm_run_dw(const _Float16* X, const CbrP& p, _Float16* Y, int B, int C,
                        int Hin, int Hout, int stride, hipStream_t stream) {
    int total = B * C * Hout * Hout;
    ssdm_dwconv<<<(total + 255) / 256, 256, 0, stream>>>(X, p.w, p.s, p.b, Y, C, Hin,
                                                         Hout, stride, total);
}

extern "C" void kernel_launch(void* const* d_in, const int* in_sizes, int n_in,
                              void* d_out, int out_size, void* d_ws, size_t ws_size,
                              hipStream_t stream) {
    auto F = [&](int i) { return (const float*)d_in[i]; };
    // jax tree_leaves sorts dict keys -> high_feat (bigger) first; otherwise insertion order.
    const bool sorted_order = (in_sizes[0] > in_sizes[1]);

    NetP net{};
    if (sorted_order) {
        net.high = F(0); net.low = F(1);
        int i = 2;
        for (int j = 0; j < 6; ++j) {  // cls: dw{b,s,w}, pw_b, pw_w
            net.cls[j].dw = {F(i + 2), F(i + 1), F(i + 0)};
            net.cls[j].pw_b = F(i + 3); net.cls[j].pw_w = F(i + 4); i += 5;
        }
        for (int j = 0; j < 4; ++j) {  // extraN: dw{b,s,w}, pw{b,s,w}
            net.edw[j] = {F(i + 2), F(i + 1), F(i + 0)};
            net.epw[j] = {F(i + 5), F(i + 4), F(i + 3)}; i += 6;
        }
        net.proj_high = {F(i + 2), F(i + 1), F(i + 0)}; i += 3;
        net.proj_low  = {F(i + 2), F(i + 1), F(i + 0)}; i += 3;
        for (int j = 0; j < 6; ++j) {
            net.reg[j].dw = {F(i + 2), F(i + 1), F(i + 0)};
            net.reg[j].pw_b = F(i + 3); net.reg[j].pw_w = F(i + 4); i += 5;
        }
    } else {
        net.low = F(0); net.high = F(1);
        int i = 2;
        net.proj_low  = {F(i), F(i + 1), F(i + 2)}; i += 3;
        net.proj_high = {F(i), F(i + 1), F(i + 2)}; i += 3;
        for (int j = 0; j < 4; ++j) {
            net.edw[j] = {F(i), F(i + 1), F(i + 2)};
            net.epw[j] = {F(i + 3), F(i + 4), F(i + 5)}; i += 6;
        }
        for (int j = 0; j < 6; ++j) {
            net.cls[j].dw = {F(i), F(i + 1), F(i + 2)};
            net.cls[j].pw_w = F(i + 3); net.cls[j].pw_b = F(i + 4); i += 5;
        }
        for (int j = 0; j < 6; ++j) {
            net.reg[j].dw = {F(i), F(i + 1), F(i + 2)};
            net.reg[j].pw_w = F(i + 3); net.reg[j].pw_b = F(i + 4); i += 5;
        }
    }

    const int B = in_sizes[sorted_order ? 1 : 0] / (80 * 20 * 20);
    const int A = 2424;
    float* out = (float*)d_out;

    // workspace layout (f16 halves)
    _Float16* ws = (_Float16*)d_ws;
    size_t o = 0;
    _Float16* p3  = ws + o; o += (size_t)B * 256 * 400;
    _Float16* p4  = ws + o; o += (size_t)B * 512 * 100;
    _Float16* p5  = ws + o; o += (size_t)B * 512 * 25;
    _Float16* p6  = ws + o; o += (size_t)B * 256 * 9;
    _Float16* p7  = ws + o; o += (size_t)B * 256 * 4;
    _Float16* p8  = ws + o; o += (size_t)B * 128 * 1;
    _Float16* dwt = ws + o; o += (size_t)B * 256 * 400;
    (void)ws_size; (void)n_in; (void)out_size;

    // ---- backbone projections (1x1 conv GEMMs), f32 input -> f16 feature
    ssdm_run_gemm<float>(net.low,  net.proj_low.w,  net.proj_low.s,  net.proj_low.b,
                         p3, nullptr, B * 400, 256, 80,  400, 0, 0, 0, 0, 0, A, stream);
    ssdm_run_gemm<float>(net.high, net.proj_high.w, net.proj_high.s, net.proj_high.b,
                         p4, nullptr, B * 100, 512, 960, 100, 0, 0, 0, 0, 0, A, stream);

    // ---- extras: dw(stride2) + pw, all f16
    const int eCi[4]   = {512, 512, 256, 256};
    const int eCo[4]   = {512, 256, 256, 128};
    const int eHin[4]  = {10, 5, 3, 2};
    const int eHout[4] = {5, 3, 2, 1};
    const _Float16* eIn[4]  = {p4, p5, p6, p7};
    _Float16*       eOut[4] = {p5, p6, p7, p8};
    for (int j = 0; j < 4; ++j) {
        ssdm_run_dw(eIn[j], net.edw[j], dwt, B, eCi[j], eHin[j], eHout[j], 2, stream);
        int HWo = eHout[j] * eHout[j];
        ssdm_run_gemm<_Float16>(dwt, net.epw[j].w, net.epw[j].s, net.epw[j].b,
                                eOut[j], nullptr, B * HWo, eCo[j], eCi[j], HWo, 0,
                                0, 0, 0, 0, A, stream);
    }

    // ---- heads: dw(stride1) + 1x1 conv scattered straight into d_out
    const _Float16* feats[6] = {p3, p4, p5, p6, p7, p8};
    const int fC[6]    = {256, 512, 512, 256, 256, 128};
    const int fH[6]    = {20, 10, 5, 3, 2, 1};
    const int fNA[6]   = {4, 6, 6, 6, 4, 4};
    const int fAoff[6] = {0, 1600, 2200, 2350, 2404, 2420};
    for (int k = 0; k < 6; ++k) {
        int HW = fH[k] * fH[k];
        // cls -> columns 4..94
        ssdm_run_dw(feats[k], net.cls[k].dw, dwt, B, fC[k], fH[k], fH[k], 1, stream);
        ssdm_run_gemm<_Float16>(dwt, net.cls[k].pw_w, nullptr, net.cls[k].pw_b,
                                nullptr, out, B * HW, fNA[k] * 91, fC[k], HW, 1,
                                fNA[k], 91, fAoff[k], 4, A, stream);
        // reg -> columns 0..3
        ssdm_run_dw(feats[k], net.reg[k].dw, dwt, B, fC[k], fH[k], fH[k], 1, stream);
        ssdm_run_gemm<_Float16>(dwt, net.reg[k].pw_w, nullptr, net.reg[k].pw_b,
                                nullptr, out, B * HW, fNA[k] * 4, fC[k], HW, 1,
                                fNA[k], 4, fAoff[k], 0, A, stream);
    }

    // ---- in-place prior decode + softmax + threshold
    int total = B * A;
    ssdm_decode<<<(total + 255) / 256, 256, 0, stream>>>(out, B, A);
}
